// WLNet_60189671686663
// MI455X (gfx1250) — matrix operand order repaired and checked by reading, not compiled, not run on hardware
//
#include <hip/hip_runtime.h>

#define B_     64
#define N_     120
#define NB_    10
#define NBOND_ 250
#define AF_    82
#define BF_    6
#define H_     300
#define HP_    304      // H padded to 19*16
#define CT_    19       // column tiles of 16
#define KA1_   96       // AF padded for K%32==0
#define KH_    320      // H padded for K%32==0
#define KU_    640      // 2H padded (two 320 halves)
#define KB2_   32       // BF padded
#define BN_    (B_ * N_)  // 7680 rows
#define MTN_   4        // (b,n) pairs per block in k_nei/k_fnei
#define MTG_   4        // 16-row M-tiles per block in k_fc1/k_self
#define MTA_   2        // 16-row M-tiles per block in k_atom

typedef __attribute__((ext_vector_type(16))) __bf16 v16bf;
typedef __attribute__((ext_vector_type(8)))  float  v8f;

union FragAB { v16bf v; uint4 q[2]; };

__device__ __forceinline__ unsigned short f2bf(float f) {
  unsigned int u = __float_as_uint(f);
  u += 0x7FFFu + ((u >> 16) & 1u);   // round-to-nearest-even
  return (unsigned short)(u >> 16);
}

// A lane frag: elems 0..7 -> k = k0 + half*8 + j ; elems 8..15 -> k0+16+half*8+j
__device__ __forceinline__ v16bf load_afrag(const unsigned short* arow, int k0) {
  FragAB a;
  a.q[0] = *(const uint4*)(arow + k0);
  a.q[1] = *(const uint4*)(arow + k0 + 16);
  return a.v;
}
// B lane frag (weights pre-transposed [n][k]): elems 0..7 -> k = k0 + half*8 + j ; elems 8..15 -> k0+16+half*8+j
__device__ __forceinline__ v16bf load_bfrag(const unsigned short* brow, int k0) {
  FragAB b;
  b.q[0] = *(const uint4*)(brow + k0);
  b.q[1] = *(const uint4*)(brow + k0 + 16);
  return b.v;
}

#define WMMA_BF16(a, b, c) \
  __builtin_amdgcn_wmma_f32_16x16x32_bf16(false, (a), false, (b), (short)0, (c), false, false)

// ---------------- prep kernels ----------------

__global__ void k_zero(unsigned int* p, int n) {
  for (int i = blockIdx.x * blockDim.x + threadIdx.x; i < n;
       i += gridDim.x * blockDim.x) p[i] = 0u;
}

__global__ void k_cvt_a1(const float* __restrict__ src, unsigned short* __restrict__ dst) {
  int i = blockIdx.x * blockDim.x + threadIdx.x;
  if (i >= BN_ * KA1_) return;
  int row = i / KA1_, k = i % KA1_;
  dst[i] = (k < AF_) ? f2bf(src[row * AF_ + k]) : (unsigned short)0;
}

__global__ void k_cvt_bond(const float* __restrict__ src, unsigned short* __restrict__ dst) {
  int i = blockIdx.x * blockDim.x + threadIdx.x;
  if (i >= B_ * NBOND_ * KB2_) return;
  int row = i / KB2_, k = i % KB2_;
  dst[i] = (k < BF_) ? f2bf(src[row * BF_ + k]) : (unsigned short)0;
}

// transpose+convert weight [krows, 300] (row-major) -> dst[n][koff + k], n<304
__global__ void k_cvt_w(const float* __restrict__ src, unsigned short* __restrict__ dst,
                        int krows, int kpad, int kspan, int koff) {
  int i = blockIdx.x * blockDim.x + threadIdx.x;
  if (i >= HP_ * kspan) return;
  int n = i / kspan, k = i % kspan;
  float v = (k < krows && n < H_) ? src[k * H_ + n] : 0.f;
  dst[n * kpad + koff + k] = f2bf(v);
}

// ---------------- fc1: atomfb = relu(atom1 @ W_fc1) ----------------

__global__ __launch_bounds__(128) void k_fc1(const unsigned short* __restrict__ a1b,
                                             const unsigned short* __restrict__ wt,
                                             unsigned short* __restrict__ atomfb) {
  __shared__ __align__(16) unsigned short lsA[MTG_][16 * 104];  // 52 dw stride (4*13 odd)
  const int tile0 = blockIdx.x * MTG_;
  for (int idx = threadIdx.x; idx < MTG_ * 16 * 48; idx += 128) {
    int p = idx / (16 * 48), rem = idx % (16 * 48);
    int m = rem / 48, j = rem % 48;
    ((unsigned int*)lsA[p])[m * 52 + j] =
        ((const unsigned int*)(a1b + ((tile0 + p) * 16 + m) * KA1_))[j];
  }
  __syncthreads();
  const int wave = threadIdx.x >> 5, lane = threadIdx.x & 31;
  const int half = lane >> 4, l16 = lane & 15;
  for (int ct = wave; ct < CT_; ct += 4) {
    const unsigned short* brow = wt + (ct * 16 + l16) * KA1_ + half * 8;
    v16bf bf[3];
#pragma unroll
    for (int ks = 0; ks < 3; ++ks) bf[ks] = load_bfrag(brow, ks * 32);
    const int col = ct * 16 + l16;
#pragma unroll
    for (int p = 0; p < MTG_; ++p) {
      const unsigned short* arow = &lsA[p][0] + l16 * 104 + half * 8;
      v8f acc = {};
#pragma unroll
      for (int ks = 0; ks < 3; ++ks) acc = WMMA_BF16(load_afrag(arow, ks * 32), bf[ks], acc);
#pragma unroll
      for (int v = 0; v < 8; ++v) {
        int m = v + half * 8;
        float val = acc[v]; val = val > 0.f ? val : 0.f;
        atomfb[((tile0 + p) * 16 + m) * KH_ + col] = f2bf(val);
      }
    }
  }
}

// --- iterated neighbor GEMM + mask + NB-reduction (MTN_ (b,n) pairs / block) ---

__global__ __launch_bounds__(128) void k_nei(const unsigned short* __restrict__ atomfb,
                                             const unsigned short* __restrict__ bondb,
                                             const long long* __restrict__ agraph,
                                             const long long* __restrict__ bgraph,
                                             const unsigned char* __restrict__ maskn,
                                             const float* __restrict__ bias,
                                             const unsigned short* __restrict__ wt,
                                             unsigned short* __restrict__ neib) {
  __shared__ __align__(16) unsigned short lsA[MTN_][16 * 328];  // 164 dw stride
  __shared__ int ia[MTN_][16], ib[MTN_][16];
  const int blk0 = blockIdx.x * MTN_;
  {
    int p = threadIdx.x >> 4, m = threadIdx.x & 15;
    if (p < MTN_) {
      int blk = blk0 + p;
      ia[p][m] = (m < NB_) ? (int)agraph[blk * NB_ + m] : -1;
      ib[p][m] = (m < NB_) ? (int)bgraph[blk * NB_ + m] : -1;
    }
  }
  __syncthreads();
  // A tiles: rows m<10 = [atom_feats[ag] (k<300) | bond_feats[bg] (k in [300,320))]
  for (int idx = threadIdx.x; idx < MTN_ * 16 * 160; idx += 128) {
    int p = idx / 2560, rem = idx % 2560;
    int m = rem / 160, j = rem % 160;
    int b = (blk0 + p) / N_;
    unsigned int val = 0u;
    int iam = ia[p][m];
    if (iam >= 0) {
      if (j < 150) val = ((const unsigned int*)(atomfb + (b * N_ + iam) * KH_))[j];
      else         val = ((const unsigned int*)(bondb + (b * NBOND_ + ib[p][m]) * KB2_))[j - 150];
    }
    ((unsigned int*)lsA[p])[m * 164 + j] = val;
  }
  __syncthreads();
  const int wave = threadIdx.x >> 5, lane = threadIdx.x & 31;
  const int half = lane >> 4, l16 = lane & 15;
  for (int ct = wave; ct < CT_; ct += 4) {
    const unsigned short* brow = wt + (ct * 16 + l16) * KH_ + half * 8;
    if (ct + 4 < CT_)
      __builtin_prefetch(wt + ((ct + 4) * 16 + l16) * KH_, 0, 0);
    v16bf bf[10];
#pragma unroll
    for (int ks = 0; ks < 10; ++ks) bf[ks] = load_bfrag(brow, ks * 32);
    const int col = ct * 16 + l16;
    const float bv = (col < H_) ? bias[col] : 0.f;
#pragma unroll
    for (int p = 0; p < MTN_; ++p) {
      const unsigned short* arow = &lsA[p][0] + l16 * 328 + half * 8;
      v8f acc = {};
#pragma unroll
      for (int ks = 0; ks < 10; ++ks) acc = WMMA_BF16(load_afrag(arow, ks * 32), bf[ks], acc);
      const int blk = blk0 + p;
      float partial = 0.f;
#pragma unroll
      for (int v = 0; v < 8; ++v) {
        int m = v + half * 8;
        float val = acc[v] + bv; val = val > 0.f ? val : 0.f;   // relu BEFORE mask
        bool keep = (m < NB_) && (maskn[blk * NB_ + m] != 0);
        partial += keep ? val : 0.f;
      }
      float total = partial + __shfl_xor(partial, 16, 32);      // combine M halves
      if (half == 0) neib[blk * KH_ + col] = f2bf(total);
    }
  }
}

// ------- atom update: atomfb_out = relu([atomfb | neib] @ W_gc_atom + b) -------

__global__ __launch_bounds__(128) void k_atom(const unsigned short* __restrict__ atomfb,
                                              const unsigned short* __restrict__ neib,
                                              const float* __restrict__ bias,
                                              const unsigned short* __restrict__ wt,
                                              unsigned short* __restrict__ atomfb_out) {
  __shared__ __align__(16) unsigned short lsA[MTA_][16 * 648];  // 324 dw stride
  const int tile0 = blockIdx.x * MTA_;
  for (int idx = threadIdx.x; idx < MTA_ * 16 * 320; idx += 128) {
    int p = idx / 5120, rem = idx % 5120;
    int m = rem / 320, j = rem % 320;
    int row = (tile0 + p) * 16 + m;
    unsigned int val = (j < 160)
        ? ((const unsigned int*)(atomfb + row * KH_))[j]
        : ((const unsigned int*)(neib + row * KH_))[j - 160];
    ((unsigned int*)lsA[p])[m * 324 + j] = val;
  }
  __syncthreads();
  const int wave = threadIdx.x >> 5, lane = threadIdx.x & 31;
  const int half = lane >> 4, l16 = lane & 15;
  for (int ct = wave; ct < CT_; ct += 4) {
    const unsigned short* brow = wt + (ct * 16 + l16) * KU_ + half * 8;
    if (ct + 4 < CT_)
      __builtin_prefetch(wt + ((ct + 4) * 16 + l16) * KU_, 0, 0);
    v8f acc[MTA_] = {};
#pragma unroll
    for (int kc = 0; kc < 2; ++kc) {       // K = 640 in two 320 chunks
      v16bf bf[10];
#pragma unroll
      for (int ks = 0; ks < 10; ++ks) bf[ks] = load_bfrag(brow, kc * 320 + ks * 32);
#pragma unroll
      for (int p = 0; p < MTA_; ++p) {
        const unsigned short* arow = &lsA[p][0] + l16 * 648 + half * 8 + kc * 320;
#pragma unroll
        for (int ks = 0; ks < 10; ++ks)
          acc[p] = WMMA_BF16(load_afrag(arow, ks * 32), bf[ks], acc[p]);
      }
    }
    const int col = ct * 16 + l16;
    const float bv = (col < H_) ? bias[col] : 0.f;
#pragma unroll
    for (int p = 0; p < MTA_; ++p) {
#pragma unroll
      for (int v = 0; v < 8; ++v) {
        int m = v + half * 8;
        float val = acc[p][v] + bv; val = val > 0.f ? val : 0.f;
        atomfb_out[((tile0 + p) * 16 + m) * KH_ + col] = f2bf(val);
      }
    }
  }
}

// ------- final: nei = sum_m mask * (atomnei@W_fc2a) * (bondnei@W_fc2b) -------

__global__ __launch_bounds__(128) void k_fnei(const unsigned short* __restrict__ atomfb,
                                              const unsigned short* __restrict__ bondb,
                                              const long long* __restrict__ agraph,
                                              const long long* __restrict__ bgraph,
                                              const unsigned char* __restrict__ maskn,
                                              const unsigned short* __restrict__ w2at,
                                              const unsigned short* __restrict__ w2bt,
                                              float* __restrict__ neif) {
  __shared__ __align__(16) unsigned short lsA[MTN_][16 * 328];
  __shared__ __align__(16) unsigned short lsB[MTN_][16 * 40];   // 20 dw stride (4*5 odd)
  __shared__ int ia[MTN_][16], ib[MTN_][16];
  const int blk0 = blockIdx.x * MTN_;
  {
    int p = threadIdx.x >> 4, m = threadIdx.x & 15;
    if (p < MTN_) {
      int blk = blk0 + p;
      ia[p][m] = (m < NB_) ? (int)agraph[blk * NB_ + m] : -1;
      ib[p][m] = (m < NB_) ? (int)bgraph[blk * NB_ + m] : -1;
    }
  }
  __syncthreads();
  for (int idx = threadIdx.x; idx < MTN_ * 16 * 160; idx += 128) {
    int p = idx / 2560, rem = idx % 2560;
    int m = rem / 160, j = rem % 160;
    int b = (blk0 + p) / N_;
    int iam = ia[p][m];
    unsigned int val = (iam >= 0)
        ? ((const unsigned int*)(atomfb + (b * N_ + iam) * KH_))[j] : 0u;
    ((unsigned int*)lsA[p])[m * 164 + j] = val;
  }
  for (int idx = threadIdx.x; idx < MTN_ * 16 * 20; idx += 128) {
    int p = idx / 320, rem = idx % 320;
    int m = rem / 20, j = rem % 20;
    int b = (blk0 + p) / N_;
    int ibm = ib[p][m];
    unsigned int val = (ibm >= 0 && j < 16)
        ? ((const unsigned int*)(bondb + (b * NBOND_ + ibm) * KB2_))[j] : 0u;
    ((unsigned int*)lsB[p])[m * 20 + j] = val;
  }
  __syncthreads();
  const int wave = threadIdx.x >> 5, lane = threadIdx.x & 31;
  const int half = lane >> 4, l16 = lane & 15;
  for (int ct = wave; ct < CT_; ct += 4) {
    const unsigned short* browA = w2at + (ct * 16 + l16) * KH_ + half * 8;
    const unsigned short* browB = w2bt + (ct * 16 + l16) * KB2_ + half * 8;
    v16bf bfA[10];
#pragma unroll
    for (int ks = 0; ks < 10; ++ks) bfA[ks] = load_bfrag(browA, ks * 32);
    v16bf bfB0 = load_bfrag(browB, 0);
    const int col = ct * 16 + l16;
#pragma unroll
    for (int p = 0; p < MTN_; ++p) {
      const unsigned short* arow = &lsA[p][0] + l16 * 328 + half * 8;
      const unsigned short* brow2 = &lsB[p][0] + l16 * 40 + half * 8;
      v8f accA = {}, accB = {};
#pragma unroll
      for (int ks = 0; ks < 10; ++ks) accA = WMMA_BF16(load_afrag(arow, ks * 32), bfA[ks], accA);
      accB = WMMA_BF16(load_afrag(brow2, 0), bfB0, accB);
      const int blk = blk0 + p;
      float partial = 0.f;
#pragma unroll
      for (int v = 0; v < 8; ++v) {
        int m = v + half * 8;
        float prod = accA[v] * accB[v];
        bool keep = (m < NB_) && (maskn[blk * NB_ + m] != 0);
        partial += keep ? prod : 0.f;
      }
      float total = partial + __shfl_xor(partial, 16, 32);
      if (half == 0) neif[blk * HP_ + col] = total;
    }
  }
}

// ------- output: out = mask_atoms * (atomfb @ W_fc2) * nei -------

__global__ __launch_bounds__(128) void k_self(const unsigned short* __restrict__ atomfb,
                                              const unsigned short* __restrict__ wt,
                                              const float* __restrict__ neif,
                                              const unsigned char* __restrict__ maska,
                                              float* __restrict__ out) {
  __shared__ __align__(16) unsigned short lsA[MTG_][16 * 328];
  const int tile0 = blockIdx.x * MTG_;
  for (int idx = threadIdx.x; idx < MTG_ * 16 * 160; idx += 128) {
    int p = idx / 2560, rem = idx % 2560;
    int m = rem / 160, j = rem % 160;
    ((unsigned int*)lsA[p])[m * 164 + j] =
        ((const unsigned int*)(atomfb + ((tile0 + p) * 16 + m) * KH_))[j];
  }
  __syncthreads();
  const int wave = threadIdx.x >> 5, lane = threadIdx.x & 31;
  const int half = lane >> 4, l16 = lane & 15;
  for (int ct = wave; ct < CT_; ct += 4) {
    const unsigned short* brow = wt + (ct * 16 + l16) * KH_ + half * 8;
    v16bf bf[10];
#pragma unroll
    for (int ks = 0; ks < 10; ++ks) bf[ks] = load_bfrag(brow, ks * 32);
    const int col = ct * 16 + l16;
#pragma unroll
    for (int p = 0; p < MTG_; ++p) {
      const unsigned short* arow = &lsA[p][0] + l16 * 328 + half * 8;
      v8f acc = {};
#pragma unroll
      for (int ks = 0; ks < 10; ++ks) acc = WMMA_BF16(load_afrag(arow, ks * 32), bf[ks], acc);
      if (col < H_) {
#pragma unroll
        for (int v = 0; v < 8; ++v) {
          int m = v + half * 8;
          int row = (tile0 + p) * 16 + m;
          float r = (maska[row] != 0) ? acc[v] * neif[row * HP_ + col] : 0.f;
          out[row * H_ + col] = r;
        }
      }
    }
  }
}

// ---------------- host ----------------

extern "C" void kernel_launch(void* const* d_in, const int* in_sizes, int n_in,
                              void* d_out, int out_size, void* d_ws, size_t ws_size,
                              hipStream_t stream) {
  const float*          atom1 = (const float*)d_in[0];
  const float*          bondf = (const float*)d_in[1];
  const long long*      ag    = (const long long*)d_in[2];
  const long long*      bg    = (const long long*)d_in[3];
  const unsigned char*  maskn = (const unsigned char*)d_in[6];
  const unsigned char*  maska = (const unsigned char*)d_in[7];
  const float* Wfc1 = (const float*)d_in[8];
  const float* Wgn  = (const float*)d_in[9];
  const float* bgn  = (const float*)d_in[10];
  const float* Wga  = (const float*)d_in[11];
  const float* bga  = (const float*)d_in[12];
  const float* W2a  = (const float*)d_in[13];
  const float* W2b  = (const float*)d_in[14];
  const float* W2   = (const float*)d_in[15];
  float* out = (float*)d_out;

  char* base = (char*)d_ws;
  size_t off = 0;
  auto alloc = [&](size_t bytes) -> void* {
    void* r = base + off;
    off = (off + bytes + 255) & ~(size_t)255;
    return r;
  };
  unsigned short* a1b   = (unsigned short*)alloc((size_t)BN_ * KA1_ * 2);
  unsigned short* bondb = (unsigned short*)alloc((size_t)B_ * NBOND_ * KB2_ * 2);
  unsigned short* wfc1t = (unsigned short*)alloc((size_t)HP_ * KA1_ * 2);
  unsigned short* wgnt  = (unsigned short*)alloc((size_t)HP_ * KH_ * 2);
  unsigned short* wgat  = (unsigned short*)alloc((size_t)HP_ * KU_ * 2);
  unsigned short* w2at  = (unsigned short*)alloc((size_t)HP_ * KH_ * 2);
  unsigned short* w2bt  = (unsigned short*)alloc((size_t)HP_ * KB2_ * 2);
  unsigned short* w2t   = (unsigned short*)alloc((size_t)HP_ * KH_ * 2);
  unsigned short* afb0  = (unsigned short*)alloc((size_t)BN_ * KH_ * 2);
  unsigned short* afb1  = (unsigned short*)alloc((size_t)BN_ * KH_ * 2);
  unsigned short* neib  = (unsigned short*)alloc((size_t)BN_ * KH_ * 2);
  float*          neif  = (float*)alloc((size_t)BN_ * HP_ * 4);

  const int actUints = BN_ * KH_ / 2;  // ushorts -> uints
  k_zero<<<1024, 256, 0, stream>>>((unsigned int*)afb0, actUints);
  k_zero<<<1024, 256, 0, stream>>>((unsigned int*)afb1, actUints);
  k_zero<<<1024, 256, 0, stream>>>((unsigned int*)neib, actUints);

  k_cvt_a1<<<(BN_ * KA1_ + 255) / 256, 256, 0, stream>>>(atom1, a1b);
  k_cvt_bond<<<(B_ * NBOND_ * KB2_ + 255) / 256, 256, 0, stream>>>(bondf, bondb);
  k_cvt_w<<<(HP_ * KA1_ + 255) / 256, 256, 0, stream>>>(Wfc1, wfc1t, AF_, KA1_, KA1_, 0);
  k_cvt_w<<<(HP_ * KH_ + 255) / 256, 256, 0, stream>>>(Wgn, wgnt, H_ + BF_, KH_, KH_, 0);
  k_cvt_w<<<(HP_ * KH_ + 255) / 256, 256, 0, stream>>>(Wga, wgat, H_, KU_, KH_, 0);
  k_cvt_w<<<(HP_ * KH_ + 255) / 256, 256, 0, stream>>>(Wga + (size_t)H_ * H_, wgat, H_, KU_, KH_, KH_);
  k_cvt_w<<<(HP_ * KH_ + 255) / 256, 256, 0, stream>>>(W2a, w2at, H_, KH_, KH_, 0);
  k_cvt_w<<<(HP_ * KB2_ + 255) / 256, 256, 0, stream>>>(W2b, w2bt, BF_, KB2_, KB2_, 0);
  k_cvt_w<<<(HP_ * KH_ + 255) / 256, 256, 0, stream>>>(W2, w2t, H_, KH_, KH_, 0);

  k_fc1<<<BN_ / 16 / MTG_, 128, 0, stream>>>(a1b, wfc1t, afb0);

  // depth iterations 0,1 (ping-pong activations)
  k_nei<<<BN_ / MTN_, 128, 0, stream>>>(afb0, bondb, ag, bg, maskn, bgn, wgnt, neib);
  k_atom<<<BN_ / 16 / MTA_, 128, 0, stream>>>(afb0, neib, bga, wgat, afb1);
  k_nei<<<BN_ / MTN_, 128, 0, stream>>>(afb1, bondb, ag, bg, maskn, bgn, wgnt, neib);
  k_atom<<<BN_ / 16 / MTA_, 128, 0, stream>>>(afb1, neib, bga, wgat, afb0);

  // final layer
  k_fnei<<<BN_ / MTN_, 128, 0, stream>>>(afb0, bondb, ag, bg, maskn, w2at, w2bt, neif);
  k_self<<<BN_ / 16 / MTG_, 128, 0, stream>>>(afb0, w2t, neif, maska, out);
}
